// GRU_78443282694242
// MI455X (gfx1250) — compile-verified, matrix operand order checked by reading, not run on hardware
//
#include <hip/hip_runtime.h>

typedef _Float16 h16v __attribute__((ext_vector_type(16)));
typedef _Float16 h8v  __attribute__((ext_vector_type(8)));
typedef float    f8v  __attribute__((ext_vector_type(8)));

#define TLEN 200
#define HID  32
#define DM   64
#define WIN  20
#define VSTR 112   // vT_s row stride (>= 96+16 pad so K-padded reads stay in-bounds)

__device__ __forceinline__ h16v pack16(h8v lo, h8v hi) {
  h16v r;
#pragma unroll
  for (int i = 0; i < 8; ++i) { r[i] = lo[i]; r[i + 8] = hi[i]; }
  return r;
}

__device__ __forceinline__ h8v cvt8(f8v a) {
  h8v r;
#pragma unroll
  for (int i = 0; i < 8; ++i) r[i] = (_Float16)a[i];
  return r;
}

// A/B operand slice: halves [p..p+7] and [p+16..p+23] (16B-aligned each)
__device__ __forceinline__ h16v lda(const _Float16* p) {
  h8v lo = *(const h8v*)(p);
  h8v hi = *(const h8v*)(p + 16);
  return pack16(lo, hi);
}

// Same, but source is f32 (weights): floats [p..p+7] and [p+16..p+23]
__device__ __forceinline__ h16v ldw(const float* p) {
  return pack16(cvt8(*(const f8v*)(p)), cvt8(*(const f8v*)(p + 16)));
}

__device__ __forceinline__ f8v wmma16(h16v a, h16v b, f8v c) {
  return __builtin_amdgcn_wmma_f32_16x16x32_f16(false, a, false, b, (short)0, c,
                                                false, false);
}

// Branch-free fast transcendentals on the serial GRU path:
// v_rcp_f32 + v_exp_f32 instead of div_scale/div_fixup chains and libm tanhf.
__device__ __forceinline__ float fast_rcp(float x) {
  return __builtin_amdgcn_rcpf(x);
}
__device__ __forceinline__ float fast_sigmoid(float x) {
  return fast_rcp(1.0f + __expf(-x));
}
__device__ __forceinline__ float fast_tanh(float x) {
  // tanh(x) = 1 - 2/(exp(2x)+1); saturates correctly for large |x|
  return 1.0f - 2.0f * fast_rcp(1.0f + __expf(2.0f * x));
}

// ---------------------------------------------------------------------------
// GRU: one wave per (16-batch tile, direction). Hidden matmul via WMMA f16.
// Writes h in f16: [B, T, 64] with cols 0..31 = fwd, 32..63 = bwd.
// ---------------------------------------------------------------------------
__global__ __launch_bounds__(32) void gru_kernel(
    const float* __restrict__ x,
    const float* __restrict__ wih_f, const float* __restrict__ whh_f,
    const float* __restrict__ bih_f, const float* __restrict__ bhh_f,
    const float* __restrict__ wih_b, const float* __restrict__ whh_b,
    const float* __restrict__ bih_b, const float* __restrict__ bhh_b,
    _Float16* __restrict__ hout) {
  const int tile = blockIdx.x >> 1;
  const int dir  = blockIdx.x & 1;  // 0 = fwd, 1 = bwd
  const int b0   = tile * 16;

  const float* wih = dir ? wih_b : wih_f;
  const float* whh = dir ? whh_b : whh_f;
  const float* bih = dir ? bih_b : bih_f;
  const float* bhh = dir ? bhh_b : bhh_f;

  __shared__ __align__(16) _Float16 h_s[16][32];
  __shared__ float x_s[16][4];

  const int lane = threadIdx.x;
  const int lh   = lane & 15;
  const int hi   = lane >> 4;
  const int k0   = hi ? 8 : 0;

  // whh [96,32] -> 6 B-operand tiles (B[k][n] = whh[ct*16+n][k])
  h16v wB[6];
#pragma unroll
  for (int ct = 0; ct < 6; ++ct) {
    const float* wr = whh + (size_t)(ct * 16 + lh) * HID;
    wB[ct] = ldw(wr + k0);
  }
  // per-lane input-projection constants for this lane's 6 columns
  float wi0[6], wi1[6], wi2[6], bsum[6], bhn[2];
#pragma unroll
  for (int ct = 0; ct < 6; ++ct) {
    int c = ct * 16 + lh;
    wi0[ct]  = wih[c * 3 + 0];
    wi1[ct]  = wih[c * 3 + 1];
    wi2[ct]  = wih[c * 3 + 2];
    bsum[ct] = bih[c] + ((ct < 4) ? bhh[c] : 0.0f);  // bhh_n kept separate
  }
  bhn[0] = bhh[64 + lh];
  bhn[1] = bhh[80 + lh];

  float hold[2][8];
#pragma unroll
  for (int a = 0; a < 2; ++a)
#pragma unroll
    for (int r = 0; r < 8; ++r) hold[a][r] = 0.0f;

  {  // zero initial hidden state in LDS
    _Float16* p = &h_s[0][0];
    for (int i = lane; i < 16 * 32; i += 32) p[i] = (_Float16)0.0f;
  }
  __syncthreads();

  for (int tt = 0; tt < TLEN; ++tt) {
    const int t = dir ? (TLEN - 1 - tt) : tt;
    if (lane < 16) {
      const float* xp = x + ((size_t)(b0 + lane) * TLEN + t) * 3;
      x_s[lane][0] = xp[0];
      x_s[lane][1] = xp[1];
      x_s[lane][2] = xp[2];
      __builtin_prefetch(xp + (dir ? -3 : 3), 0, 1);
    }
    __syncthreads();

    h16v A = lda(&h_s[lh][k0]);  // h tile, A layout, K=0..31
    f8v d[6];
#pragma unroll
    for (int ct = 0; ct < 6; ++ct) {
      f8v c = {};
      d[ct] = wmma16(A, wB[ct], c);  // gh = h @ whh.T (per 16-col tile)
    }
    __syncthreads();  // everyone done reading h_s before overwrite

#pragma unroll
    for (int ct = 0; ct < 2; ++ct) {
#pragma unroll
      for (int r = 0; r < 8; ++r) {
        const int m = r + 8 * hi;  // batch row in tile (C/D layout)
        const float x0 = x_s[m][0], x1 = x_s[m][1], x2 = x_s[m][2];
        float gr = bsum[ct]     + x0 * wi0[ct]     + x1 * wi1[ct]     + x2 * wi2[ct]     + d[ct][r];
        float gz = bsum[ct + 2] + x0 * wi0[ct + 2] + x1 * wi1[ct + 2] + x2 * wi2[ct + 2] + d[ct + 2][r];
        float gi_n = bsum[ct + 4] + x0 * wi0[ct + 4] + x1 * wi1[ct + 4] + x2 * wi2[ct + 4];
        float rr = fast_sigmoid(gr);
        float zz = fast_sigmoid(gz);
        float nn = fast_tanh(gi_n + rr * (d[ct + 4][r] + bhn[ct]));
        float hv = (1.0f - zz) * nn + zz * hold[ct][r];
        hold[ct][r] = hv;
        const int j = ct * 16 + lh;  // hidden index 0..31
        h_s[m][j] = (_Float16)hv;
        hout[((size_t)(b0 + m) * TLEN + t) * DM + dir * HID + j] = (_Float16)hv;
      }
    }
    __syncthreads();
  }
}

// ---------------------------------------------------------------------------
// Fused banded attention + FC. Block = (b, dir, group of 64 targets),
// 4 waves; wave w owns target tile w. All matmuls via WMMA f16->f32.
// ---------------------------------------------------------------------------
__global__ __launch_bounds__(128) void attn_kernel(
    const _Float16* __restrict__ hbuf,
    const float* __restrict__ af_wq, const float* __restrict__ af_bq,
    const float* __restrict__ af_wk, const float* __restrict__ af_bk,
    const float* __restrict__ af_wv, const float* __restrict__ af_bv,
    const float* __restrict__ ab_wq, const float* __restrict__ ab_bq,
    const float* __restrict__ ab_wk, const float* __restrict__ ab_bk,
    const float* __restrict__ ab_wv, const float* __restrict__ ab_bv,
    const float* __restrict__ fcw, const float* __restrict__ fcb,
    float* __restrict__ out) {
  const int blk = blockIdx.x;
  const int b   = blk >> 3;
  const int dir = (blk >> 2) & 1;
  const int grp = blk & 3;
  const int gt0 = grp * 64;                      // first target of group
  const int cstart = gt0 + (dir ? 0 : -WIN);     // first staged context row

  const float* wq = dir ? ab_wq : af_wq;
  const float* bq = dir ? ab_bq : af_bq;
  const float* wk = dir ? ab_wk : af_wk;
  const float* bk = dir ? ab_bk : af_bk;
  const float* wv = dir ? ab_wv : af_wv;
  const float* bv = dir ? ab_bv : af_bv;

  __shared__ __align__(16) _Float16 q_s[96][64];     // context q
  __shared__ __align__(16) _Float16 k_s[64][64];     // target k
  __shared__ __align__(16) _Float16 vT_s[64][VSTR];  // context v, transposed [d][ctx]
  __shared__ __align__(16) _Float16 a_buf[4][16][64];
  __shared__ float s_buf[4][16][48];

  const int tid  = threadIdx.x;
  const int wave = tid >> 5;
  const int lane = tid & 31;
  const int lh   = lane & 15;
  const int hi   = lane >> 4;
  const int k0   = hi ? 8 : 0;

  // zero the K-pad columns of vT_s (ctx 96..111) so padded AV reads are clean
  for (int i = tid; i < 64 * 16; i += 128)
    vT_s[i >> 4][96 + (i & 15)] = (_Float16)0.0f;

  // ---------------- projections: 64 jobs of (16x16 tile, K=64) ----------------
  for (int it = 0; it < 16; ++it) {
    const int jw = it * 4 + wave;
    int proj, tile, ctl;
    if (jw < 24)      { proj = 0; tile = jw >> 2;        ctl = jw & 3; }  // q (ctx)
    else if (jw < 48) { proj = 1; tile = (jw - 24) >> 2; ctl = jw & 3; }  // v (ctx)
    else              { proj = 2; tile = (jw - 48) >> 2; ctl = jw & 3; }  // k (tgt)

    const float* w  = (proj == 0) ? wq : (proj == 1) ? wv : wk;
    const float* bs = (proj == 0) ? bq : (proj == 1) ? bv : bk;

    int row = (proj == 2) ? (gt0 + tile * 16 + lh) : (cstart + tile * 16 + lh);
    row = min(max(row, 0), TLEN - 1);  // clamp; invalid rows masked later
    const _Float16* arow = hbuf + ((size_t)b * TLEN + row) * DM;
    h16v A0 = lda(arow + k0);        // K 0..31
    h16v A1 = lda(arow + 32 + k0);   // K 32..63
    const float* wrow = w + (size_t)(ctl * 16 + lh) * DM;
    h16v B0 = ldw(wrow + k0);
    h16v B1 = ldw(wrow + 32 + k0);
    f8v c = {};
    c = wmma16(A0, B0, c);
    c = wmma16(A1, B1, c);
    const float bb = bs[ctl * 16 + lh];
#pragma unroll
    for (int r = 0; r < 8; ++r) {
      const int M = r + 8 * hi;
      const int col = ctl * 16 + lh;
      const _Float16 v = (_Float16)(c[r] + bb);
      if (proj == 0)      q_s[tile * 16 + M][col] = v;
      else if (proj == 1) vT_s[col][tile * 16 + M] = v;
      else                k_s[tile * 16 + M][col] = v;
    }
  }
  __syncthreads();

  // ---------------- scores: s[i][jrel] = k_t[i] . q_c[j] / 8 ----------------
  const int w = wave;  // target tile owned by this wave
  {
    h16v Ak0 = lda(&k_s[w * 16 + lh][k0]);
    h16v Ak1 = lda(&k_s[w * 16 + lh][32 + k0]);
#pragma unroll
    for (int ctt = 0; ctt < 3; ++ctt) {
      const _Float16* qrow = &q_s[(w + ctt) * 16 + lh][0];
      h16v B0 = lda(qrow + k0);
      h16v B1 = lda(qrow + 32 + k0);
      f8v c = {};
      c = wmma16(Ak0, B0, c);
      c = wmma16(Ak1, B1, c);
#pragma unroll
      for (int r = 0; r < 8; ++r) {
        const int M = r + 8 * hi;
        const int jrel = ctt * 16 + lh;             // offset in 48-wide window
        const int i = gt0 + w * 16 + M;
        const int j = cstart + w * 16 + jrel;
        const bool valid = (jrel >= M) && (jrel <= M + WIN) && (j >= 0) &&
                           (j < TLEN) && (i < TLEN);
        s_buf[w][M][jrel] = valid ? c[r] * 0.125f : -1e30f;
      }
    }
  }
  __syncthreads();

  // ---------------- softmax per target row ----------------
  if (lane < 16) {
    float mx = -1e30f;
    for (int jj = 0; jj < 48; ++jj) mx = fmaxf(mx, s_buf[w][lane][jj]);
    float sum = 0.0f;
    for (int jj = 0; jj < 48; ++jj) sum += __expf(s_buf[w][lane][jj] - mx);
    const float inv = fast_rcp(sum);
    for (int jj = 0; jj < 48; ++jj)
      a_buf[w][lane][jj] = (_Float16)(__expf(s_buf[w][lane][jj] - mx) * inv);
    for (int jj = 48; jj < 64; ++jj) a_buf[w][lane][jj] = (_Float16)0.0f;
  }
  __syncthreads();

  // ---------------- O = A @ V  (K = 48 padded to 64) ----------------
  f8v o[4];
  {
    h16v Aa0 = lda(&a_buf[w][lh][k0]);
    h16v Aa1 = lda(&a_buf[w][lh][32 + k0]);
#pragma unroll
    for (int dt = 0; dt < 4; ++dt) {
      const _Float16* vrow = &vT_s[dt * 16 + lh][w * 16];
      h16v B0 = lda(vrow + k0);
      h16v B1 = lda(vrow + 32 + k0);
      f8v c = {};
      c = wmma16(Aa0, B0, c);
      c = wmma16(Aa1, B1, c);
      o[dt] = c;
    }
  }

  // ---------------- FC: out = O . fc_w + fc_b ----------------
  float fw[4];
#pragma unroll
  for (int dt = 0; dt < 4; ++dt) fw[dt] = fcw[dt * 16 + lh];
  float p[8];
#pragma unroll
  for (int r = 0; r < 8; ++r)
    p[r] = o[0][r] * fw[0] + o[1][r] * fw[1] + o[2][r] * fw[2] + o[3][r] * fw[3];
#pragma unroll
  for (int m = 1; m < 16; m <<= 1)
#pragma unroll
    for (int r = 0; r < 8; ++r) p[r] += __shfl_xor(p[r], m, 16);
  if (lh == 0) {
    const float bo = fcb[0];
#pragma unroll
    for (int r = 0; r < 8; ++r) {
      const int i = gt0 + w * 16 + r + 8 * hi;
      if (i < TLEN) out[((size_t)b * TLEN + i) * 2 + dir] = p[r] + bo;
    }
  }
}

// ---------------------------------------------------------------------------
extern "C" void kernel_launch(void* const* d_in, const int* in_sizes, int n_in,
                              void* d_out, int out_size, void* d_ws,
                              size_t ws_size, hipStream_t stream) {
  (void)in_sizes; (void)n_in; (void)out_size; (void)ws_size;
  const float* x        = (const float*)d_in[0];
  const float* gwih_f   = (const float*)d_in[1];
  const float* gwhh_f   = (const float*)d_in[2];
  const float* gbih_f   = (const float*)d_in[3];
  const float* gbhh_f   = (const float*)d_in[4];
  const float* gwih_b   = (const float*)d_in[5];
  const float* gwhh_b   = (const float*)d_in[6];
  const float* gbih_b   = (const float*)d_in[7];
  const float* gbhh_b   = (const float*)d_in[8];
  const float* af_wq    = (const float*)d_in[9];
  const float* af_bq    = (const float*)d_in[10];
  const float* af_wk    = (const float*)d_in[11];
  const float* af_bk    = (const float*)d_in[12];
  const float* af_wv    = (const float*)d_in[13];
  const float* af_bv    = (const float*)d_in[14];
  const float* ab_wq    = (const float*)d_in[15];
  const float* ab_bq    = (const float*)d_in[16];
  const float* ab_wk    = (const float*)d_in[17];
  const float* ab_bk    = (const float*)d_in[18];
  const float* ab_wv    = (const float*)d_in[19];
  const float* ab_bv    = (const float*)d_in[20];
  const float* fc_w     = (const float*)d_in[21];
  const float* fc_b     = (const float*)d_in[22];

  _Float16* hbuf = (_Float16*)d_ws;  // [1024][200][64] f16 = 26.2 MB

  gru_kernel<<<128, 32, 0, stream>>>(x, gwih_f, gwhh_f, gbih_f, gbhh_f,
                                     gwih_b, gwhh_b, gbih_b, gbhh_b, hbuf);

  attn_kernel<<<1024 * 8, 128, 0, stream>>>(
      hbuf, af_wq, af_bq, af_wk, af_bk, af_wv, af_bv, ab_wq, ab_bq, ab_wk,
      ab_bk, ab_wv, ab_bv, fc_w, fc_b, (float*)d_out);
}
